// RTNoEdgeInit_79706003079220
// MI455X (gfx1250) — compile-verified, matrix operand order checked by reading, not run on hardware
//
#include <hip/hip_runtime.h>

#define Bb 4
#define Nn 256
#define Dd 128
#define Ee 128
#define Hh 8
#define Ll 3

typedef __attribute__((ext_vector_type(16))) _Float16 v16h;
typedef __attribute__((ext_vector_type(8)))  _Float16 v8h;
typedef __attribute__((ext_vector_type(8)))  float    v8f;

__device__ __forceinline__ v8f wmma_f16(v16h a, v16h b, v8f c) {
  return __builtin_amdgcn_wmma_f32_16x16x32_f16(false, a, false, b, (short)0, c,
                                                false, false);
}

union Frag16 { v16h v; v8h h[2]; };

// Per-lane base pointer of a 16x32 (MxK) f16 fragment. ISA layout:
// lanes 0-15: M=lane, K halves {0..7,16..23}; lanes 16-31: M=lane-16,
// K halves {8..15,24..31}. Two 16-byte loads per lane. The B-operand
// (KxN, N striped over lanes) has the identical lane pattern when the
// matrix is stored transposed (NxK row-major), so one helper serves both.
__device__ __forceinline__ const _Float16* frag_base(const _Float16* base, int ld) {
  int lane = threadIdx.x & 31;
  return base + (size_t)(lane & 15) * ld + (lane >> 4) * 8;
}
__device__ __forceinline__ v16h frag_load(const _Float16* lb) {
  Frag16 f;
  f.h[0] = *(const v8h*)(lb);
  f.h[1] = *(const v8h*)(lb + 16);
  return f.v;
}
__device__ __forceinline__ void st4h(_Float16* d, float4 v) {
  d[0] = (_Float16)v.x; d[1] = (_Float16)v.y;
  d[2] = (_Float16)v.z; d[3] = (_Float16)v.w;
}

// ---------------- weight convert: W (L,K,Nc) f32 -> WT (L,Nc,K) f16 ---------
__global__ void wcvt_kernel(const float* __restrict__ W, _Float16* __restrict__ WT,
                            int K, int Nc) {
  size_t idx = (size_t)blockIdx.x * 256 + threadIdx.x;
  size_t per = (size_t)K * Nc;
  if (idx >= (size_t)Ll * per) return;
  size_t l = idx / per, rem = idx % per;
  size_t k = rem / Nc, nc = rem % Nc;
  WT[l * per + nc * (size_t)K + k] = (_Float16)W[idx];
}

// ---------------- node qkv: qkv = n @ Wqkv_n + b  (1024 x 384) --------------
__global__ void node_qkv_kernel(const float* __restrict__ n,
                                const _Float16* __restrict__ WT,  // 384 x 128
                                const float* __restrict__ bias,
                                float* __restrict__ qkv) {
  int idx = blockIdx.x * 256 + threadIdx.x;  // B*N*384
  int row = idx / 384, c = idx % 384;
  const float* nr = n + (size_t)row * Dd;
  const _Float16* wr = WT + (size_t)c * Dd;
  float a = bias[c];
#pragma unroll 8
  for (int k = 0; k < Dd; ++k) a += nr[k] * (float)wr[k];
  qkv[idx] = a;
}

// ---------------- fused edge-biased attention per (b,i) ---------------------
// scores = ||q_n+q_e||^2 (k overwritten by q in ref), softmax_j,
// out = v_n + sum_j att*v_e ; then @Wo + bo + residual + LN1.
#define ATTN_LDS (Nn*136*2 + Hh*Nn*4 + Nn*132*4 + 128*4 + 128*4)
__global__ __launch_bounds__(256) void attn_kernel(
    const float* __restrict__ e, const float* __restrict__ qkv,
    float* __restrict__ n_io,
    const _Float16* __restrict__ WqkveT,  // 384 x 128
    const float* __restrict__ bqkve,
    const _Float16* __restrict__ WoT,     // 128 x 128
    const float* __restrict__ bo,
    const float* __restrict__ lng, const float* __restrict__ lnb_) {
  extern __shared__ char smem[];
  _Float16* et  = (_Float16*)smem;                       // [256][136] f16
  float* sc     = (float*)(smem + Nn * 136 * 2);         // [8][256]
  float* V      = sc + Hh * Nn;                          // [256][132]
  float* outb   = V + Nn * 132;                          // [128]
  float* lnbuf  = outb + 128;                            // [128]

  int bi = blockIdx.x;
  int b = bi >> 8, i = bi & 255;
  int t = threadIdx.x, lane = t & 31, w = t >> 5;
  size_t row = (size_t)(b * Nn + i);
  const float* erow = e + row * Nn * Ee;

  // stage e[b,i,:,:] as f16
  for (int k = 0; k < 32; ++k) {
    int j = w * 32 + k;
    float4 v = ((const float4*)(erow + (size_t)j * Ee))[lane];
    st4h(et + j * 136 + lane * 4, v);
  }
  __syncthreads();

  const float* qrow = qkv + row * 384;
  int lr = lane & 15, hi = lane >> 4;

  // ---- phase 1: scores for head h = w.
  // Transposed GEMM: D[d][j] = (Wq^T x e^T) so each lane owns one column j
  // with d = r + 8*hi across the 8 acc regs -> in-register reduction plus a
  // single shfl_xor(16) instead of a ds_bpermute chain.
  {
    int h = w;
    float qn_r[8];
#pragma unroll
    for (int r = 0; r < 8; ++r)
      qn_r[r] = qrow[h * 48 + r + 8 * hi] + bqkve[h * 48 + r + 8 * hi];
    const _Float16* bp = frag_base(WqkveT + (size_t)(h * 48) * Dd, Dd);
    v16h wf[4];
#pragma unroll
    for (int ks = 0; ks < 4; ++ks) wf[ks] = frag_load(bp + ks * 32);
    const _Float16* ebase0 = frag_base(et, 136);
    v16h ef[4];
#pragma unroll
    for (int ks = 0; ks < 4; ++ks) ef[ks] = frag_load(ebase0 + ks * 32);
#pragma unroll 1
    for (int jt = 0; jt < 16; ++jt) {
      v16h efn[4];
      if (jt < 15) {
        const _Float16* ap = ebase0 + (size_t)(jt + 1) * 16 * 136;
#pragma unroll
        for (int ks = 0; ks < 4; ++ks) efn[ks] = frag_load(ap + ks * 32);
      } else {
#pragma unroll
        for (int ks = 0; ks < 4; ++ks) efn[ks] = ef[ks];
      }
      v8f acc = {0, 0, 0, 0, 0, 0, 0, 0};
#pragma unroll
      for (int ks = 0; ks < 4; ++ks) acc = wmma_f16(wf[ks], ef[ks], acc);
      float s = 0.f;
#pragma unroll
      for (int r = 0; r < 8; ++r) {
        float q = acc[r] + qn_r[r];
        s = fmaf(q, q, s);
      }
      s += __shfl_xor(s, 16, 32);  // combine the two d-halves of column j
      if (hi == 0) sc[h * Nn + jt * 16 + lr] = s * 0.25f;  // 1/sqrt(16)
#pragma unroll
      for (int ks = 0; ks < 4; ++ks) ef[ks] = efn[ks];
    }
  }
  __syncthreads();

  // ---- softmax over j, head h = w ----
  {
    int h = w;
    float vv[8], mx = -1e30f;
#pragma unroll
    for (int u = 0; u < 8; ++u) { vv[u] = sc[h * Nn + u * 32 + lane]; mx = fmaxf(mx, vv[u]); }
    for (int m = 1; m < 32; m <<= 1) mx = fmaxf(mx, __shfl_xor(mx, m, 32));
    float sum = 0.f;
#pragma unroll
    for (int u = 0; u < 8; ++u) { vv[u] = __expf(vv[u] - mx); sum += vv[u]; }
    for (int m = 1; m < 32; m <<= 1) sum += __shfl_xor(sum, m, 32);
    float inv = 1.f / sum;
#pragma unroll
    for (int u = 0; u < 8; ++u) sc[h * Nn + u * 32 + lane] = vv[u] * inv;
  }
  __syncthreads();

  // ---- phase 2: V_e = e_tile @ Wv  -> LDS ----
#pragma unroll 1
  for (int jj = 0; jj < 2; ++jj) {
    int jt = w * 2 + jj;
    const _Float16* ap = frag_base(et + jt * 16 * 136, 136);
    v16h af[4];
#pragma unroll
    for (int ks = 0; ks < 4; ++ks) af[ks] = frag_load(ap + ks * 32);
#pragma unroll 1
    for (int h2 = 0; h2 < Hh; ++h2) {
      v8f acc = {0, 0, 0, 0, 0, 0, 0, 0};
      const _Float16* bp = frag_base(WqkveT + (size_t)(h2 * 48 + 32) * Dd, Dd);
#pragma unroll
      for (int ks = 0; ks < 4; ++ks) acc = wmma_f16(af[ks], frag_load(bp + ks * 32), acc);
      float bv = bqkve[h2 * 48 + 32 + lr];
#pragma unroll
      for (int r = 0; r < 8; ++r)
        V[(size_t)(jt * 16 + r + 8 * hi) * 132 + h2 * 16 + lr] = acc[r] + bv;
    }
  }
  __syncthreads();

  // ---- phase 3: out = v_n + att @ V_e ----
  if (t < 128) {
    int h3 = t >> 4, d = t & 15;
    float acc = qrow[h3 * 48 + 32 + d];  // sum(att)=1 picks up v_n exactly once
    const float4* att4 = (const float4*)(sc + h3 * Nn);
#pragma unroll 4
    for (int j4 = 0; j4 < 64; ++j4) {
      float4 a4 = att4[j4];
      const float* vp = V + (size_t)(j4 * 4) * 132 + t;
      acc += a4.x * vp[0] + a4.y * vp[132] + a4.z * vp[264] + a4.w * vp[396];
    }
    outb[t] = acc;
  }
  __syncthreads();
  // ---- phase 4: Wo projection + residual + LN1 ----
  if (t < 128) {
    float acc = bo[t] + n_io[row * Dd + t];
    const _Float16* wr = WoT + (size_t)t * Dd;
#pragma unroll 8
    for (int k = 0; k < Dd; ++k) acc += outb[k] * (float)wr[k];
    lnbuf[t] = acc;
  }
  __syncthreads();
  if (t < 128) {
    float s = 0.f, s2 = 0.f;
    for (int k = 0; k < Dd; ++k) { float x = lnbuf[k]; s += x; s2 += x * x; }
    float m = s * (1.f / Dd);
    float var = s2 * (1.f / Dd) - m * m;
    float rs = rsqrtf(var + 1e-5f);
    n_io[row * Dd + t] = (lnbuf[t] - m) * rs * lng[t] + lnb_[t];
  }
}

// ---------------- node FFN + LN2 (per row) ----------------------------------
__global__ __launch_bounds__(256) void node_ffn_kernel(
    float* __restrict__ n_io,
    const _Float16* __restrict__ Wn1T,  // 512 x 128
    const float* __restrict__ bn1,
    const _Float16* __restrict__ Wn2T,  // 128 x 512
    const float* __restrict__ bn2,
    const float* __restrict__ lng, const float* __restrict__ lnb_) {
  __shared__ float nrow[128];
  __shared__ float h1[512];
  __shared__ float lnbuf[128];
  size_t row = blockIdx.x;
  int t = threadIdx.x;
  if (t < 128) nrow[t] = n_io[row * Dd + t];
  __syncthreads();
  for (int c = t; c < 512; c += 256) {
    float a = bn1[c];
    const _Float16* wr = Wn1T + (size_t)c * Dd;
#pragma unroll 8
    for (int k = 0; k < Dd; ++k) a += nrow[k] * (float)wr[k];
    h1[c] = fmaxf(a, 0.f);
  }
  __syncthreads();
  if (t < 128) {
    float a = bn2[t];
    const _Float16* wr = Wn2T + (size_t)t * 512;
#pragma unroll 8
    for (int k = 0; k < 512; ++k) a += h1[k] * (float)wr[k];
    lnbuf[t] = a + nrow[t];
  }
  __syncthreads();
  if (t < 128) {
    float s = 0.f, s2 = 0.f;
    for (int k = 0; k < Dd; ++k) { float x = lnbuf[k]; s += x; s2 += x * x; }
    float m = s * (1.f / Dd);
    float rs = rsqrtf(s2 * (1.f / Dd) - m * m + 1e-5f);
    n_io[row * Dd + t] = (lnbuf[t] - m) * rs * lng[t] + lnb_[t];
  }
}

// Shared epilogue: outb[64][132] += e_resid row, LayerNorm, vectorized write.
// 4 threads per row; partial sums reduced through LDS.
__device__ __forceinline__ void edge_ln_epilogue(
    const float* __restrict__ e_res, float* __restrict__ e_dst, size_t ebase,
    int jt, float* outb, float* part, float* mrs,
    const float* __restrict__ lng, const float* __restrict__ lnb_, int t) {
  int row = t >> 2, seg = t & 3;
  int j = jt * 64 + row;
  {
    const float4* eo = (const float4*)(e_res + ebase + (size_t)j * Ee + seg * 32);
    float* orow = outb + (size_t)row * 132 + seg * 32;
    float s = 0.f, s2 = 0.f;
#pragma unroll
    for (int u = 0; u < 8; ++u) {
      float4 v = eo[u];
      float* o = orow + u * 4;
      float y0 = v.x + o[0], y1 = v.y + o[1], y2 = v.z + o[2], y3 = v.w + o[3];
      o[0] = y0; o[1] = y1; o[2] = y2; o[3] = y3;
      s += y0 + y1 + y2 + y3;
      s2 += y0 * y0 + y1 * y1 + y2 * y2 + y3 * y3;
    }
    part[(row * 4 + seg) * 2] = s;
    part[(row * 4 + seg) * 2 + 1] = s2;
  }
  __syncthreads();
  if (t < 64) {
    float s = part[t * 8] + part[t * 8 + 2] + part[t * 8 + 4] + part[t * 8 + 6];
    float s2 = part[t * 8 + 1] + part[t * 8 + 3] + part[t * 8 + 5] + part[t * 8 + 7];
    float m = s * (1.f / Ee);
    mrs[t * 2] = m;
    mrs[t * 2 + 1] = rsqrtf(s2 * (1.f / Ee) - m * m + 1e-5f);
  }
  __syncthreads();
  {
    float m = mrs[row * 2], rs = mrs[row * 2 + 1];
    float4* dst = (float4*)(e_dst + ebase + (size_t)j * Ee + seg * 32);
    const float* orow = outb + (size_t)row * 132 + seg * 32;
#pragma unroll
    for (int u = 0; u < 8; ++u) {
      int c = seg * 32 + u * 4;
      float4 y;
      y.x = (orow[u * 4 + 0] - m) * rs * lng[c + 0] + lnb_[c + 0];
      y.y = (orow[u * 4 + 1] - m) * rs * lng[c + 1] + lnb_[c + 1];
      y.z = (orow[u * 4 + 2] - m) * rs * lng[c + 2] + lnb_[c + 2];
      y.w = (orow[u * 4 + 3] - m) * rs * lng[c + 3] + lnb_[c + 3];
      dst[u] = y;
    }
  }
}

// ---------------- fused edge MLP1: relu(cat@W1+b1)@W2+b2, +e, LN ------------
// cat = [e_ij, e_ji, n_j, n_i] (512) ; block = 64 edges of row i.
#define MLP1_LDS (64*520*2 + 64*132*4 + 512*4 + 128*4)
__global__ __launch_bounds__(256) void edge_mlp1_kernel(
    const float* __restrict__ e_in, float* __restrict__ e_out,
    const float* __restrict__ n_cur,
    const _Float16* __restrict__ W1T,  // 512 x 512
    const float* __restrict__ b1,
    const _Float16* __restrict__ W2T,  // 128 x 512
    const float* __restrict__ b2,
    const float* __restrict__ lng, const float* __restrict__ lnb_) {
  extern __shared__ char smem[];
  _Float16* cat = (_Float16*)smem;               // [64][520] ; reused as h1
  float* outb = (float*)(smem + 64 * 520 * 2);   // [64][132]
  float* part = outb + 64 * 132;                 // [64][4][2]
  float* mrs = part + 512;                       // [64][2]
  int blk = blockIdx.x;
  int jt = blk & 3, i = (blk >> 2) & 255, b = blk >> 10;
  int t = threadIdx.x, lane = t & 31, w = t >> 5;
  int lr = lane & 15, hi = lane >> 4;
  size_t ebase = (size_t)(b * Nn + i) * Nn * Ee;

  // build cat tile
  for (int k = 0; k < 8; ++k) {
    int r = w * 8 + k;
    int j = jt * 64 + r;
    _Float16* d = cat + (size_t)r * 520;
    float4 v;
    v = ((const float4*)(e_in + ebase + (size_t)j * Ee))[lane];
    st4h(d + lane * 4, v);
    v = ((const float4*)(e_in + ((size_t)(b * Nn + j) * Nn + i) * Ee))[lane];
    st4h(d + 128 + lane * 4, v);
    v = ((const float4*)(n_cur + (size_t)(b * Nn + j) * Dd))[lane];
    st4h(d + 256 + lane * 4, v);
    v = ((const float4*)(n_cur + (size_t)(b * Nn + i) * Dd))[lane];
    st4h(d + 384 + lane * 4, v);
  }
  __syncthreads();

  int rt = w & 3, cg = w >> 2;  // rows rt*16.., cols cg*256..
  v8f acc[16];
  v8f z = {0, 0, 0, 0, 0, 0, 0, 0};
#pragma unroll
  for (int q = 0; q < 16; ++q) acc[q] = z;

  // GEMM1: 64x512 = cat(64x512) @ W1 (A-frag double-buffered)
  {
    const _Float16* abase = frag_base(cat + rt * 16 * 520, 520);
    v16h a_cur = frag_load(abase);
#pragma unroll 1
    for (int ks = 0; ks < 16; ++ks) {
      v16h a_nxt = a_cur;
      if (ks < 15) a_nxt = frag_load(abase + (ks + 1) * 32);
#pragma unroll
      for (int tt = 0; tt < 16; ++tt) {
        const _Float16* bb =
            frag_base(W1T + (size_t)(cg * 256 + tt * 16) * 512, 512) + ks * 32;
        acc[tt] = wmma_f16(a_cur, frag_load(bb), acc[tt]);
      }
      a_cur = a_nxt;
    }
  }
  __syncthreads();  // all waves done reading cat
  // bias + relu -> f16 h1 (reuse cat storage)
#pragma unroll
  for (int tt = 0; tt < 16; ++tt) {
    float bv = b1[cg * 256 + tt * 16 + lr];
#pragma unroll
    for (int r = 0; r < 8; ++r) {
      float x = fmaxf(acc[tt][r] + bv, 0.f);
      cat[(size_t)(rt * 16 + r + 8 * hi) * 520 + cg * 256 + tt * 16 + lr] = (_Float16)x;
    }
  }
  __syncthreads();

  // GEMM2: 64x128 = h1(64x512) @ W2
  v8f a2[4];
#pragma unroll
  for (int q = 0; q < 4; ++q) a2[q] = z;
  {
    const _Float16* abase = frag_base(cat + rt * 16 * 520, 520);
    v16h a_cur = frag_load(abase);
#pragma unroll 1
    for (int ks = 0; ks < 16; ++ks) {
      v16h a_nxt = a_cur;
      if (ks < 15) a_nxt = frag_load(abase + (ks + 1) * 32);
#pragma unroll
      for (int tt = 0; tt < 4; ++tt) {
        const _Float16* bb =
            frag_base(W2T + (size_t)(cg * 64 + tt * 16) * 512, 512) + ks * 32;
        a2[tt] = wmma_f16(a_cur, frag_load(bb), a2[tt]);
      }
      a_cur = a_nxt;
    }
  }
#pragma unroll
  for (int tt = 0; tt < 4; ++tt) {
    float bv = b2[cg * 64 + tt * 16 + lr];
#pragma unroll
    for (int r = 0; r < 8; ++r)
      outb[(size_t)(rt * 16 + r + 8 * hi) * 132 + cg * 64 + tt * 16 + lr] =
          a2[tt][r] + bv;
  }
  __syncthreads();

  edge_ln_epilogue(e_in, e_out, ebase, jt, outb, part, mrs, lng, lnb_, t);
}

// ---------------- fused edge MLP2: relu(e@W1+b1)@W2+b2, +e, LN (in place) ---
#define MLP2_LDS (64*136*2 + 64*520*2 + 64*132*4 + 512*4 + 128*4)
__global__ __launch_bounds__(256) void edge_mlp2_kernel(
    float* __restrict__ e_io,
    const _Float16* __restrict__ W1T,  // 512 x 128
    const float* __restrict__ b1,
    const _Float16* __restrict__ W2T,  // 128 x 512
    const float* __restrict__ b2,
    const float* __restrict__ lng, const float* __restrict__ lnb_) {
  extern __shared__ char smem[];
  _Float16* ein = (_Float16*)smem;                            // [64][136]
  _Float16* h1 = (_Float16*)(smem + 64 * 136 * 2);            // [64][520]
  float* outb = (float*)(smem + 64 * 136 * 2 + 64 * 520 * 2); // [64][132]
  float* part = outb + 64 * 132;                              // [64][4][2]
  float* mrs = part + 512;                                    // [64][2]
  int blk = blockIdx.x;
  int jt = blk & 3, i = (blk >> 2) & 255, b = blk >> 10;
  int t = threadIdx.x, lane = t & 31, w = t >> 5;
  int lr = lane & 15, hi = lane >> 4;
  size_t ebase = (size_t)(b * Nn + i) * Nn * Ee;

  for (int k = 0; k < 8; ++k) {
    int r = w * 8 + k;
    float4 v = ((const float4*)(e_io + ebase + (size_t)(jt * 64 + r) * Ee))[lane];
    st4h(ein + (size_t)r * 136 + lane * 4, v);
  }
  __syncthreads();

  int rt = w & 3, cg = w >> 2;
  v8f z = {0, 0, 0, 0, 0, 0, 0, 0};
  v8f acc[16];
#pragma unroll
  for (int q = 0; q < 16; ++q) acc[q] = z;

  // GEMM1: 64x512 = e(64x128) @ W1
  v16h af[4];
#pragma unroll
  for (int ks = 0; ks < 4; ++ks)
    af[ks] = frag_load(frag_base(ein + rt * 16 * 136, 136) + ks * 32);
#pragma unroll 1
  for (int ks = 0; ks < 4; ++ks) {
#pragma unroll
    for (int tt = 0; tt < 16; ++tt) {
      const _Float16* bb =
          frag_base(W1T + (size_t)(cg * 256 + tt * 16) * Dd, Dd) + ks * 32;
      acc[tt] = wmma_f16(af[ks], frag_load(bb), acc[tt]);
    }
  }
#pragma unroll
  for (int tt = 0; tt < 16; ++tt) {
    float bv = b1[cg * 256 + tt * 16 + lr];
#pragma unroll
    for (int r = 0; r < 8; ++r) {
      float x = fmaxf(acc[tt][r] + bv, 0.f);
      h1[(size_t)(rt * 16 + r + 8 * hi) * 520 + cg * 256 + tt * 16 + lr] = (_Float16)x;
    }
  }
  __syncthreads();

  // GEMM2: 64x128 = h1(64x512) @ W2
  v8f a2[4];
#pragma unroll
  for (int q = 0; q < 4; ++q) a2[q] = z;
  {
    const _Float16* abase = frag_base(h1 + rt * 16 * 520, 520);
    v16h a_cur = frag_load(abase);
#pragma unroll 1
    for (int ks = 0; ks < 16; ++ks) {
      v16h a_nxt = a_cur;
      if (ks < 15) a_nxt = frag_load(abase + (ks + 1) * 32);
#pragma unroll
      for (int tt = 0; tt < 4; ++tt) {
        const _Float16* bb =
            frag_base(W2T + (size_t)(cg * 64 + tt * 16) * 512, 512) + ks * 32;
        a2[tt] = wmma_f16(a_cur, frag_load(bb), a2[tt]);
      }
      a_cur = a_nxt;
    }
  }
#pragma unroll
  for (int tt = 0; tt < 4; ++tt) {
    float bv = b2[cg * 64 + tt * 16 + lr];
#pragma unroll
    for (int r = 0; r < 8; ++r)
      outb[(size_t)(rt * 16 + r + 8 * hi) * 132 + cg * 64 + tt * 16 + lr] =
          a2[tt][r] + bv;
  }
  __syncthreads();

  edge_ln_epilogue(e_io, e_io, ebase, jt, outb, part, mrs, lng, lnb_, t);
}

// ---------------------------------------------------------------------------
extern "C" void kernel_launch(void* const* d_in, const int* in_sizes, int n_in,
                              void* d_out, int out_size, void* d_ws, size_t ws_size,
                              hipStream_t stream) {
  const float* node_f = (const float*)d_in[0];
  const float* edge_f = (const float*)d_in[1];
  const float* Wqkv_n = (const float*)d_in[2];
  const float* bqkv_n = (const float*)d_in[3];
  const float* Wqkv_e = (const float*)d_in[4];
  const float* bqkv_e = (const float*)d_in[5];
  const float* Wo     = (const float*)d_in[6];
  const float* bo     = (const float*)d_in[7];
  const float* Wn1    = (const float*)d_in[8];
  const float* bn1    = (const float*)d_in[9];
  const float* Wn2    = (const float*)d_in[10];
  const float* bn2    = (const float*)d_in[11];
  const float* ln1n_g = (const float*)d_in[12];
  const float* ln1n_b = (const float*)d_in[13];
  const float* ln2n_g = (const float*)d_in[14];
  const float* ln2n_b = (const float*)d_in[15];
  const float* We1a   = (const float*)d_in[16];
  const float* be1a   = (const float*)d_in[17];
  const float* We1b   = (const float*)d_in[18];
  const float* be1b   = (const float*)d_in[19];
  const float* We2a   = (const float*)d_in[20];
  const float* be2a   = (const float*)d_in[21];
  const float* We2b   = (const float*)d_in[22];
  const float* be2b   = (const float*)d_in[23];
  const float* ln1e_g = (const float*)d_in[24];
  const float* ln1e_b = (const float*)d_in[25];
  const float* ln2e_g = (const float*)d_in[26];
  const float* ln2e_b = (const float*)d_in[27];

  char* ws = (char*)d_ws;
  size_t o = 0;
  auto take = [&](size_t bytes) { char* p = ws + o; o = (o + bytes + 255) & ~(size_t)255; return p; };
  _Float16* WqkvnT = (_Float16*)take((size_t)Ll * 384 * 128 * 2);
  _Float16* WqkveT = (_Float16*)take((size_t)Ll * 384 * 128 * 2);
  _Float16* WoT    = (_Float16*)take((size_t)Ll * 128 * 128 * 2);
  _Float16* Wn1T   = (_Float16*)take((size_t)Ll * 512 * 128 * 2);
  _Float16* Wn2T   = (_Float16*)take((size_t)Ll * 128 * 512 * 2);
  _Float16* We1aT  = (_Float16*)take((size_t)Ll * 512 * 512 * 2);
  _Float16* We1bT  = (_Float16*)take((size_t)Ll * 128 * 512 * 2);
  _Float16* We2aT  = (_Float16*)take((size_t)Ll * 512 * 128 * 2);
  _Float16* We2bT  = (_Float16*)take((size_t)Ll * 128 * 512 * 2);
  float* qkvbuf = (float*)take((size_t)Bb * Nn * 384 * 4);
  float* n_cur  = (float*)take((size_t)Bb * Nn * Dd * 4);
  float* e_a    = (float*)take((size_t)Bb * Nn * Nn * Ee * 4);
  float* e_b    = (float*)take((size_t)Bb * Nn * Nn * Ee * 4);

  hipFuncSetAttribute((const void*)attn_kernel,
                      hipFuncAttributeMaxDynamicSharedMemorySize, ATTN_LDS);
  hipFuncSetAttribute((const void*)edge_mlp1_kernel,
                      hipFuncAttributeMaxDynamicSharedMemorySize, MLP1_LDS);
  hipFuncSetAttribute((const void*)edge_mlp2_kernel,
                      hipFuncAttributeMaxDynamicSharedMemorySize, MLP2_LDS);

  auto cvt = [&](const float* W, _Float16* WT, int K, int Nc) {
    size_t n = (size_t)Ll * K * Nc;
    wcvt_kernel<<<(unsigned)((n + 255) / 256), 256, 0, stream>>>(W, WT, K, Nc);
  };
  cvt(Wqkv_n, WqkvnT, 128, 384);
  cvt(Wqkv_e, WqkveT, 128, 384);
  cvt(Wo,     WoT,    128, 128);
  cvt(Wn1,    Wn1T,   128, 512);
  cvt(Wn2,    Wn2T,   512, 128);
  cvt(We1a,   We1aT,  512, 512);
  cvt(We1b,   We1bT,  512, 128);
  cvt(We2a,   We2aT,  128, 512);
  cvt(We2b,   We2bT,  512, 128);

  hipMemcpyAsync(n_cur, node_f, (size_t)Bb * Nn * Dd * 4,
                 hipMemcpyDeviceToDevice, stream);
  hipMemcpyAsync(e_a, edge_f, (size_t)Bb * Nn * Nn * Ee * 4,
                 hipMemcpyDeviceToDevice, stream);

  float* e_cur = e_a;
  float* e_nxt = e_b;
  for (int l = 0; l < Ll; ++l) {
    node_qkv_kernel<<<(Bb * Nn * 384) / 256, 256, 0, stream>>>(
        n_cur, WqkvnT + (size_t)l * 384 * 128, bqkv_n + l * 384, qkvbuf);
    attn_kernel<<<Bb * Nn, 256, ATTN_LDS, stream>>>(
        e_cur, qkvbuf, n_cur, WqkveT + (size_t)l * 384 * 128, bqkv_e + l * 384,
        WoT + (size_t)l * 128 * 128, bo + l * 128,
        ln1n_g + l * 128, ln1n_b + l * 128);
    node_ffn_kernel<<<Bb * Nn, 256, 0, stream>>>(
        n_cur, Wn1T + (size_t)l * 512 * 128, bn1 + l * 512,
        Wn2T + (size_t)l * 128 * 512, bn2 + l * 128,
        ln2n_g + l * 128, ln2n_b + l * 128);
    edge_mlp1_kernel<<<Bb * Nn * (Nn / 64), 256, MLP1_LDS, stream>>>(
        e_cur, e_nxt, n_cur, We1aT + (size_t)l * 512 * 512, be1a + l * 512,
        We1bT + (size_t)l * 128 * 512, be1b + l * 128,
        ln1e_g + l * 128, ln1e_b + l * 128);
    edge_mlp2_kernel<<<Bb * Nn * (Nn / 64), 256, MLP2_LDS, stream>>>(
        e_nxt, We2aT + (size_t)l * 512 * 128, be2a + l * 512,
        We2bT + (size_t)l * 128 * 512, be2b + l * 128,
        ln2e_g + l * 128, ln2e_b + l * 128);
    float* tmp = e_cur; e_cur = e_nxt; e_nxt = tmp;
  }
  hipMemcpyAsync(d_out, n_cur, (size_t)Bb * Nn * Dd * 4,
                 hipMemcpyDeviceToDevice, stream);
}
